// Mamba_69870527971709
// MI455X (gfx1250) — matrix operand order repaired and checked by reading, not compiled, not run on hardware
//
#include <hip/hip_runtime.h>
#include <hip/hip_bf16.h>

// ---------- types ----------
typedef __attribute__((ext_vector_type(16))) __bf16 v16bf;
typedef __attribute__((ext_vector_type(8)))  __bf16 v8bf;
typedef __attribute__((ext_vector_type(8)))  float  v8f;

// ---------- problem constants ----------
#define BATCH   2
#define LSEQ    1024
#define DMODEL  1024
#define INNER   2048          // EXPAND * D_MODEL
#define MROWS   (BATCH*LSEQ)  // 2048
#define DSTATE  16

// ---------- GEMM tiling ----------
#define BM 128
#define BN 128
#define BK 32
#define LDSS 40               // padded LDS row stride in bf16 (32 data + 8 pad)
#define ABUF_B (BM * LDSS * 2)   // bytes per A buffer
#define BBUF_B (BN * LDSS * 2)   // bytes per B buffer

// =====================================================================
// fp32 -> bf16 convert (grid-stride)
// =====================================================================
__global__ __launch_bounds__(256) void f32_to_bf16_kernel(
    const float* __restrict__ in, unsigned short* __restrict__ out, int n)
{
    int i = blockIdx.x * blockDim.x + threadIdx.x;
    int stride = gridDim.x * blockDim.x;
    for (; i < n; i += stride) {
        __bf16 b = (__bf16)in[i];
        out[i] = __builtin_bit_cast(unsigned short, b);
    }
}

// =====================================================================
// C[M,N] (f32) = A[M,K](bf16, row-major) * B[N,K](bf16, row-major)^T
// 256 threads = 8 wave32 (4x2); each wave owns a 32x64 tile (2x4 WMMA).
// K-slabs streamed via GLOBAL_LOAD_ASYNC_TO_LDS_B128, double-buffered.
// =====================================================================
__global__ __launch_bounds__(256) void gemm_bf16_wmma_kernel(
    const unsigned short* __restrict__ Araw,
    const unsigned short* __restrict__ Braw,
    float* __restrict__ C,
    int M, int N, int K)
{
    const __bf16* __restrict__ A  = reinterpret_cast<const __bf16*>(Araw);
    const __bf16* __restrict__ Bm = reinterpret_cast<const __bf16*>(Braw);

    __shared__ __align__(16) __bf16 At[2][BM * LDSS];
    __shared__ __align__(16) __bf16 Bt[2][BN * LDSS];

    const int tid   = threadIdx.x;
    const int lane  = tid & 31;
    const int wave  = tid >> 5;
    const int waveM = wave & 3;   // 4 waves along M (32 rows each)
    const int waveN = wave >> 2;  // 2 waves along N (64 cols each)
    const int g     = lane >> 4;  // half-wave id
    const int hl    = lane & 15;

    const int blockM = blockIdx.x * BM;
    const int blockN = blockIdx.y * BN;

    // ---- per-thread staging assignment: 2 A chunks + 2 B chunks (16B each)
    const int r0 = tid >> 2;            // rows 0..63 ; second chunk row +64
    const int c0 = (tid & 3) * 8;       // bf16 column offset within K-slab
    const __bf16* aSrc0 = A  + (size_t)(blockM + r0)      * K + c0;
    const __bf16* aSrc1 = A  + (size_t)(blockM + r0 + 64) * K + c0;
    const __bf16* bSrc0 = Bm + (size_t)(blockN + r0)      * K + c0;
    const __bf16* bSrc1 = Bm + (size_t)(blockN + r0 + 64) * K + c0;

    // LDS byte offsets (generic->LDS offset = low 32 bits)
    const unsigned ldsA0 = (unsigned)(unsigned long long)&At[0][r0 * LDSS + c0];
    const unsigned ldsA1 = (unsigned)(unsigned long long)&At[0][(r0 + 64) * LDSS + c0];
    const unsigned ldsB0 = (unsigned)(unsigned long long)&Bt[0][r0 * LDSS + c0];
    const unsigned ldsB1 = (unsigned)(unsigned long long)&Bt[0][(r0 + 64) * LDSS + c0];

    auto stage = [&](int buf, int kk) {
        unsigned aoff = buf ? (unsigned)ABUF_B : 0u;
        unsigned boff = buf ? (unsigned)BBUF_B : 0u;
        asm volatile("global_load_async_to_lds_b128 %0, %1, off"
                     :: "v"(ldsA0 + aoff), "v"((unsigned long long)(aSrc0 + kk)) : "memory");
        asm volatile("global_load_async_to_lds_b128 %0, %1, off"
                     :: "v"(ldsA1 + aoff), "v"((unsigned long long)(aSrc1 + kk)) : "memory");
        asm volatile("global_load_async_to_lds_b128 %0, %1, off"
                     :: "v"(ldsB0 + boff), "v"((unsigned long long)(bSrc0 + kk)) : "memory");
        asm volatile("global_load_async_to_lds_b128 %0, %1, off"
                     :: "v"(ldsB1 + boff), "v"((unsigned long long)(bSrc1 + kk)) : "memory");
    };
    #define WAIT_ASYNC() asm volatile("s_wait_asynccnt 0x0" ::: "memory")

    v8f acc[2][4];
    #pragma unroll
    for (int i = 0; i < 2; ++i)
        #pragma unroll
        for (int j = 0; j < 4; ++j) acc[i][j] = {};

    stage(0, 0);
    WAIT_ASYNC();
    __syncthreads();

    for (int kk = 0; kk < K; kk += BK) {
        const int buf = (kk >> 5) & 1;
        if (kk + BK < K) stage(buf ^ 1, kk + BK);

        // ---- build fragments per ISA 7.12.2 layouts ----
        v16bf aFrag[2], bFrag[4];
        #pragma unroll
        for (int mi = 0; mi < 2; ++mi) {
            // A: lane holds row m; elem i -> K = 16*(i/8) + 8*g + (i%8)
            int m = waveM * 32 + mi * 16 + hl;
            v8bf lo = *(const v8bf*)&At[buf][m * LDSS + g * 8];
            v8bf hi = *(const v8bf*)&At[buf][m * LDSS + 16 + g * 8];
            #pragma unroll
            for (int i = 0; i < 8; ++i) { aFrag[mi][i] = lo[i]; aFrag[mi][8 + i] = hi[i]; }
        }
        #pragma unroll
        for (int nj = 0; nj < 4; ++nj) {
            // B: lane holds col n; elem i -> K = 16*(i/8) + 8*g + (i%8)
            int n = waveN * 64 + nj * 16 + hl;
            v8bf bl = *(const v8bf*)&Bt[buf][n * LDSS + g * 8];
            v8bf bh = *(const v8bf*)&Bt[buf][n * LDSS + 16 + g * 8];
            #pragma unroll
            for (int i = 0; i < 8; ++i) { bFrag[nj][i] = bl[i]; bFrag[nj][8 + i] = bh[i]; }
        }

        #pragma unroll
        for (int mi = 0; mi < 2; ++mi)
            #pragma unroll
            for (int nj = 0; nj < 4; ++nj)
                acc[mi][nj] = __builtin_amdgcn_wmma_f32_16x16x32_bf16(
                    false, aFrag[mi], false, bFrag[nj],
                    (short)0, acc[mi][nj], false, false);

        WAIT_ASYNC();       // next slab fully landed in the other buffer
        __syncthreads();    // all waves done computing on `buf`
    }

    // ---- store: VGPR v holds M = 8*g + v, N = hl ----
    #pragma unroll
    for (int mi = 0; mi < 2; ++mi)
        #pragma unroll
        for (int nj = 0; nj < 4; ++nj) {
            int row0 = blockM + waveM * 32 + mi * 16 + g * 8;
            int col  = blockN + waveN * 64 + nj * 16 + hl;
            #pragma unroll
            for (int v = 0; v < 8; ++v)
                C[(size_t)(row0 + v) * N + col] = acc[mi][nj][v];
        }
    #undef WAIT_ASYNC
}

// =====================================================================
// depthwise causal conv(4) + sigmoid + selective scan + gate
// one thread per (batch, channel); state & params in registers.
// xp: (B, L, 2*INNER) f32 interleaved [u, gate]; out: (B, L, INNER) bf16
// =====================================================================
__global__ __launch_bounds__(256) void mamba_scan_kernel(
    const float* __restrict__ xp,
    const float* __restrict__ conv_w,   // (INNER, 1, 4)
    const float* __restrict__ conv_b,   // (INNER)
    const float* __restrict__ Aar,      // (INNER, 16)
    const float* __restrict__ Bar,      // (INNER, 16)
    const float* __restrict__ Car,      // (INNER, 16)
    unsigned short* __restrict__ yscan) // (B, L, INNER) bf16
{
    int t = blockIdx.x * blockDim.x + threadIdx.x;  // 0..4095
    int b = t >> 11;          // / INNER
    int n = t & (INNER - 1);  // % INNER

    const float cw0 = conv_w[n * 4 + 0];
    const float cw1 = conv_w[n * 4 + 1];
    const float cw2 = conv_w[n * 4 + 2];
    const float cw3 = conv_w[n * 4 + 3];
    const float cb  = conv_b[n];

    float Av[DSTATE], Bv[DSTATE], Cv[DSTATE], st[DSTATE];
    #pragma unroll
    for (int s = 0; s < DSTATE; ++s) {
        Av[s] = Aar[n * DSTATE + s];
        Bv[s] = Bar[n * DSTATE + s];
        Cv[s] = Car[n * DSTATE + s];
        st[s] = 0.0f;
    }

    float w0 = 0.f, w1 = 0.f, w2 = 0.f, w3 = 0.f;
    const float* xpb = xp + (size_t)b * LSEQ * (2 * INNER);
    unsigned short* yb = yscan + (size_t)b * LSEQ * INNER;

    for (int l = 0; l < LSEQ; ++l) {
        float u    = xpb[(size_t)l * (2 * INNER) + 2 * n];
        float gate = xpb[(size_t)l * (2 * INNER) + 2 * n + 1];
        w0 = w1; w1 = w2; w2 = w3; w3 = u;
        float xc = cw0 * w0 + cw1 * w1 + cw2 * w2 + cw3 * w3 + cb;
        float delta = 1.0f / (1.0f + __expf(-xc));
        float du = delta * xc;
        float y = 0.0f;
        #pragma unroll
        for (int s = 0; s < DSTATE; ++s) {
            st[s] = Av[s] * delta * st[s] + Bv[s] * du;
            y += st[s] * Cv[s];
        }
        float sg = 1.0f / (1.0f + __expf(-gate));
        __bf16 o = (__bf16)(y * sg);
        yb[(size_t)l * INNER + n] = __builtin_bit_cast(unsigned short, o);
    }
}

// =====================================================================
// launch
// =====================================================================
extern "C" void kernel_launch(void* const* d_in, const int* in_sizes, int n_in,
                              void* d_out, int out_size, void* d_ws, size_t ws_size,
                              hipStream_t stream) {
    const float* x      = (const float*)d_in[0];  // (2,1024,1024)
    const float* W_in   = (const float*)d_in[1];  // (4096,1024)
    const float* conv_w = (const float*)d_in[2];  // (2048,1,4)
    const float* conv_b = (const float*)d_in[3];  // (2048)
    const float* Aar    = (const float*)d_in[4];  // (2048,16)
    const float* Bar    = (const float*)d_in[5];  // (2048,16)
    const float* Car    = (const float*)d_in[6];  // (2048,16)
    // d_in[7] = D_param (unused by reference forward)
    const float* W_out  = (const float*)d_in[8];  // (1024,2048)

    char* ws = (char*)d_ws;
    size_t off = 0;
    unsigned short* Xb    = (unsigned short*)(ws + off); off += (size_t)MROWS * DMODEL * 2;       // 4 MB
    unsigned short* Winb  = (unsigned short*)(ws + off); off += (size_t)(2 * INNER) * DMODEL * 2; // 8 MB
    unsigned short* Woutb = (unsigned short*)(ws + off); off += (size_t)DMODEL * INNER * 2;       // 4 MB
    float*          xpB   = (float*)(ws + off);          off += (size_t)MROWS * (2 * INNER) * 4;  // 32 MB
    unsigned short* Yb    = (unsigned short*)(ws + off); off += (size_t)MROWS * INNER * 2;        // 8 MB
    (void)off; (void)ws_size; (void)in_sizes; (void)n_in; (void)out_size;

    // 1) downconvert operands to bf16 (bandwidth-bound, one pass)
    f32_to_bf16_kernel<<<1024, 256, 0, stream>>>(x,     Xb,    MROWS * DMODEL);
    f32_to_bf16_kernel<<<1024, 256, 0, stream>>>(W_in,  Winb,  (2 * INNER) * DMODEL);
    f32_to_bf16_kernel<<<1024, 256, 0, stream>>>(W_out, Woutb, DMODEL * INNER);

    // 2) in_proj: xp[2048,4096] = X[2048,1024] * W_in[4096,1024]^T
    gemm_bf16_wmma_kernel<<<dim3(MROWS / BM, (2 * INNER) / BN), 256, 0, stream>>>(
        Xb, Winb, xpB, MROWS, 2 * INNER, DMODEL);

    // 3) conv + selective scan + gate  -> Yb[2048,2048] bf16
    mamba_scan_kernel<<<(BATCH * INNER) / 256, 256, 0, stream>>>(
        xpB, conv_w, conv_b, Aar, Bar, Car, Yb);

    // 4) out_proj: out[2048,1024] = Y[2048,2048] * W_out[1024,2048]^T
    gemm_bf16_wmma_kernel<<<dim3(MROWS / BM, DMODEL / BN), 256, 0, stream>>>(
        Yb, Woutb, (float*)d_out, MROWS, DMODEL, INNER);
}
